// GCN_33586644255338
// MI455X (gfx1250) — matrix operand order repaired and checked by reading, not compiled, not run on hardware
//
#include <hip/hip_runtime.h>

typedef __attribute__((ext_vector_type(16))) __bf16 v16bf;
typedef __attribute__((ext_vector_type(8)))  __bf16 v8bf;
typedef __attribute__((ext_vector_type(8)))  float  v8f;

#define N_NODES 50000
#define N_EDGES 800000

__device__ __forceinline__ unsigned short f2bf(float f) {
    unsigned int u = __float_as_uint(f);
    u += 0x7FFFu + ((u >> 16) & 1u);          // round-to-nearest-even
    return (unsigned short)(u >> 16);
}

__device__ __forceinline__ float fatomic_add(float* p, float v) {
#if defined(__HIP_DEVICE_COMPILE__)
    return unsafeAtomicAdd(p, v);             // global_atomic_add_f32
#else
    return atomicAdd(p, v);
#endif
}

// ---------------- elementwise / setup kernels ----------------

__global__ void fill_zero(float* __restrict__ p, int n) {
    int i = blockIdx.x * blockDim.x + threadIdx.x;
    if (i < n) p[i] = 0.0f;
}

__global__ void deg_accum(const int* __restrict__ row, const float* __restrict__ w,
                          float* __restrict__ deg, int E) {
    int e = blockIdx.x * blockDim.x + threadIdx.x;
    if (e < E) fatomic_add(&deg[row[e]], w[e]);
}

__global__ void norm_compute(const int* __restrict__ row, const float* __restrict__ w,
                             const float* __restrict__ deg, float* __restrict__ norm, int E) {
    int e = blockIdx.x * blockDim.x + threadIdx.x;
    if (e < E) {
        float d = deg[row[e]];
        // dinv = rsqrt(d); norm = dinv * w * dinv = w / d  (d > 0 else 0)
        norm[e] = (d > 0.0f) ? (w[e] / d) : 0.0f;
    }
}

// 4-wide f32 -> bf16 (n must be a multiple of 4; all our totals are)
__global__ void cvt_bf16_v4(const float* __restrict__ src, unsigned short* __restrict__ dst,
                            int nvec) {
    int i = blockIdx.x * blockDim.x + threadIdx.x;
    if (i < nvec) {
        float4 v = reinterpret_cast<const float4*>(src)[i];
        ushort4 o;
        o.x = f2bf(v.x); o.y = f2bf(v.y); o.z = f2bf(v.z); o.w = f2bf(v.w);
        reinterpret_cast<ushort4*>(dst)[i] = o;
    }
}

// W is [K][NOUT] row-major f32 ; Wt is [NOUT][K] row-major bf16
__global__ void transpose_cvt(const float* __restrict__ W, unsigned short* __restrict__ Wt,
                              int K, int NOUT) {
    int i = blockIdx.x * blockDim.x + threadIdx.x;
    if (i < K * NOUT) {
        int n = i / K, k = i % K;
        Wt[i] = f2bf(W[(size_t)k * NOUT + n]);
    }
}

__global__ void bias_init(float* __restrict__ out, const float* __restrict__ b,
                          int total, int fmask) {
    int i = blockIdx.x * blockDim.x + threadIdx.x;
    if (i < total) out[i] = b[i & fmask];
}

// 4-wide fused relu + f32->bf16
__global__ void relu_cvt_v4(const float* __restrict__ h, unsigned short* __restrict__ hb,
                            int nvec) {
    int i = blockIdx.x * blockDim.x + threadIdx.x;
    if (i < nvec) {
        float4 v = reinterpret_cast<const float4*>(h)[i];
        ushort4 o;
        o.x = f2bf(v.x > 0.0f ? v.x : 0.0f);
        o.y = f2bf(v.y > 0.0f ? v.y : 0.0f);
        o.z = f2bf(v.z > 0.0f ? v.z : 0.0f);
        o.w = f2bf(v.w > 0.0f ? v.w : 0.0f);
        reinterpret_cast<ushort4*>(hb)[i] = o;
    }
}

// ---------------- bf16 WMMA GEMM:  C[M][NOUT] = A[M][KDIM] * Wt[NOUT][KDIM]^T ----------------
// Block: 256 threads = 8 waves. Block tile: 128 rows x 64 cols. Wave tile: 16 rows x 64 cols
// (4 x v_wmma_f32_16x16x32_bf16 per k-step, A fragment reused 4x, B panel staged in LDS).

template <int KDIM, int NOUT>
__global__ __launch_bounds__(256)
void gemm_bf16_wmma(const unsigned short* __restrict__ A,
                    const unsigned short* __restrict__ Wt,
                    float* __restrict__ C, int M) {
    __shared__ __align__(16) unsigned short sW[64 * KDIM];   // 64 output cols of Wt

    const int lane = threadIdx.x & 31;
    const int wave = threadIdx.x >> 5;
    const int half = lane >> 4;      // 0: lanes 0-15, 1: lanes 16-31
    const int l16  = lane & 15;
    const int cb   = blockIdx.y * 64;
    const int rb   = blockIdx.x * 128 + wave * 16;

    // cooperative copy: Wt rows [cb, cb+64) are contiguous -> LDS (ds_store_b128)
    {
        const uint4* src = reinterpret_cast<const uint4*>(Wt + (size_t)cb * KDIM);
        uint4* dst = reinterpret_cast<uint4*>(sW);
        const int nvec = (64 * KDIM) / 8;   // uint4 == 8 ushorts
        for (int i = threadIdx.x; i < nvec; i += 256) dst[i] = src[i];
    }
    __syncthreads();

    // A row for this lane (clamped: WMMA requires EXEC all-1s, mask at store instead)
    int arow = rb + l16;
    if (arow >= M) arow = M - 1;
    const unsigned short* Arow = A + (size_t)arow * KDIM;

    v8f acc[4] = {v8f{}, v8f{}, v8f{}, v8f{}};

    for (int k = 0; k < KDIM; k += 32) {
        // A fragment (16x32 bf16): lane half=0 -> K {k..k+7, k+16..k+23};
        //                          half=1 -> K {k+8..k+15, k+24..k+31}
        const int off = half * 8;
        v8bf alo = *reinterpret_cast<const v8bf*>(Arow + k + off);
        v8bf ahi = *reinterpret_cast<const v8bf*>(Arow + k + off + 16);
        v16bf a;
#pragma unroll
        for (int i = 0; i < 8; ++i) { a[i] = alo[i]; a[8 + i] = ahi[i]; }

#pragma unroll
        for (int t = 0; t < 4; ++t) {
            // B fragment (32x16 bf16): lane holds col N = l16 (+16*t tile),
            // K = k + 8*half + {0..7, 16..23}, from transposed weight panel in LDS
            const unsigned short* Bcol = sW + (size_t)(t * 16 + l16) * KDIM + k + half * 8;
            v8bf blo = *reinterpret_cast<const v8bf*>(Bcol);
            v8bf bhi = *reinterpret_cast<const v8bf*>(Bcol + 16);
            v16bf b;
#pragma unroll
            for (int i = 0; i < 8; ++i) { b[i] = blo[i]; b[8 + i] = bhi[i]; }

            acc[t] = __builtin_amdgcn_wmma_f32_16x16x32_bf16(
                false, a, false, b, (short)0, acc[t], false, false);
        }
    }

    // C layout: VGPR r -> row rb + r + 8*half, col cb + l16 + 16*t
#pragma unroll
    for (int r = 0; r < 8; ++r) {
        int row = rb + r + half * 8;
        if (row < M) {
            float* outp = C + (size_t)row * NOUT + cb + l16;
#pragma unroll
            for (int t = 0; t < 4; ++t) outp[t * 16] = acc[t][r];
        }
    }
}

// ---------------- edge gather-scale-scatter (4 features / thread) ----------------

template <int F>
__global__ void scatter_edges_v4(const int* __restrict__ src, const int* __restrict__ dst,
                                 const float* __restrict__ norm, const float* __restrict__ X,
                                 float* __restrict__ out, int E) {
    constexpr int FV = F / 4;                       // float4 chunks per edge
    int idx = blockIdx.x * blockDim.x + threadIdx.x;  // total = E*FV < 2^31
    if (idx < E * FV) {
        int e = idx / FV;
        int f4 = idx % FV;
        float nv = norm[e];
        const float4 xv = reinterpret_cast<const float4*>(X + (size_t)src[e] * F)[f4];
        float* op = out + (size_t)dst[e] * F + f4 * 4;
        fatomic_add(op + 0, nv * xv.x);
        fatomic_add(op + 1, nv * xv.y);
        fatomic_add(op + 2, nv * xv.z);
        fatomic_add(op + 3, nv * xv.w);
    }
}

// ---------------- launcher ----------------

extern "C" void kernel_launch(void* const* d_in, const int* in_sizes, int n_in,
                              void* d_out, int out_size, void* d_ws, size_t ws_size,
                              hipStream_t stream) {
    const float* x  = (const float*)d_in[0];
    const int*   ei = (const int*)d_in[1];     // [2][E]
    const float* ew = (const float*)d_in[2];
    const float* W1 = (const float*)d_in[3];
    const float* b1 = (const float*)d_in[4];
    const float* W2 = (const float*)d_in[5];
    const float* b2 = (const float*)d_in[6];
    float* out = (float*)d_out;

    const int N = N_NODES, E = N_EDGES;
    const int* e_src = ei;          // row
    const int* e_dst = ei + E;      // col

    // workspace carve-out (aliased where lifetimes allow)
    char* ws = (char*)d_ws;
    auto carve = [&](size_t bytes) { char* p = ws; ws += (bytes + 255) & ~(size_t)255; return p; };
    float*          deg  = (float*)carve((size_t)N * 4);
    float*          norm = (float*)carve((size_t)E * 4);
    unsigned short* Xb   = (unsigned short*)carve((size_t)N * 256 * 2);  // layer1 A; reused as Hb
    unsigned short* W1t  = (unsigned short*)carve((size_t)128 * 256 * 2);
    unsigned short* W2t  = (unsigned short*)carve((size_t)64 * 128 * 2);
    float*          xw1  = (float*)carve((size_t)N * 128 * 4);           // reused as xw2
    float*          hacc = (float*)carve((size_t)N * 128 * 4);
    unsigned short* Hb   = Xb;
    float*          xw2  = xw1;

    const int T = 256;

    // norm = w / deg[row]
    fill_zero<<<(N + T - 1) / T, T, 0, stream>>>(deg, N);
    deg_accum<<<(E + T - 1) / T, T, 0, stream>>>(e_src, ew, deg, E);
    norm_compute<<<(E + T - 1) / T, T, 0, stream>>>(e_src, ew, deg, norm, E);

    // bf16 conversions
    cvt_bf16_v4<<<((N * 64) + T - 1) / T, T, 0, stream>>>(x, Xb, N * 64);   // N*256/4
    transpose_cvt<<<((256 * 128) + T - 1) / T, T, 0, stream>>>(W1, W1t, 256, 128);
    transpose_cvt<<<((128 * 64) + T - 1) / T, T, 0, stream>>>(W2, W2t, 128, 64);

    // layer 1: xw1 = x @ W1 (WMMA), hacc = b1 + scatter(norm * xw1[src])
    dim3 g1((N + 127) / 128, 128 / 64);
    gemm_bf16_wmma<256, 128><<<g1, T, 0, stream>>>(Xb, W1t, xw1, N);
    bias_init<<<((N * 128) + T - 1) / T, T, 0, stream>>>(hacc, b1, N * 128, 127);
    scatter_edges_v4<128><<<((E * 32) + T - 1) / T, T, 0, stream>>>(e_src, e_dst, norm, xw1, hacc, E);

    // relu + cvt -> Hb (bf16)
    relu_cvt_v4<<<((N * 32) + T - 1) / T, T, 0, stream>>>(hacc, Hb, N * 32);  // N*128/4

    // layer 2: xw2 = h @ W2 (WMMA), out = b2 + scatter(norm * xw2[src])
    dim3 g2((N + 127) / 128, 64 / 64);
    gemm_bf16_wmma<128, 64><<<g2, T, 0, stream>>>(Hb, W2t, xw2, N);
    bias_init<<<((N * 64) + T - 1) / T, T, 0, stream>>>(out, b2, N * 64, 63);
    scatter_edges_v4<64><<<((E * 16) + T - 1) / T, T, 0, stream>>>(e_src, e_dst, norm, xw2, out, E);
}
